// MinGRU_22187801051584
// MI455X (gfx1250) — matrix operand order repaired and checked by reading, not compiled, not run on hardware
//
#include <hip/hip_runtime.h>
#include <stdint.h>

#define DEV __device__ __forceinline__

constexpr int B_ = 8, T_ = 8192, D_ = 512, H_ = 512;
constexpr int TC = 128;          // time rows per block (== scan chunk)
constexpr int NT = 64;           // h cols per block
constexpr int KC = 32;           // bf16 WMMA K depth
constexpr int NCHUNK = T_ / TC;  // 64
constexpr int AStride = 40;      // padded LDS row stride (bf16 elems)

typedef __attribute__((ext_vector_type(16))) __bf16 v16bf;
typedef __attribute__((ext_vector_type(8)))  float  v8f;

union FragU { v16bf v; uint4 q[2]; };

DEV unsigned short f32_to_bf16(float f) {
  unsigned u = __builtin_bit_cast(unsigned, f);
  unsigned r = 0x7FFFu + ((u >> 16) & 1u);
  return (unsigned short)((u + r) >> 16);
}
DEV unsigned pack_bf16x2(float lo, float hi) {
  return (unsigned)f32_to_bf16(lo) | ((unsigned)f32_to_bf16(hi) << 16);
}

// ---- async global->LDS copy (gfx1250), with sync fallback -------------------
#if __has_builtin(__builtin_amdgcn_global_load_async_to_lds_b128) && __has_builtin(__builtin_amdgcn_s_wait_asynccnt)
#define HAVE_ASYNC_LDS 1
#else
#define HAVE_ASYNC_LDS 0
#endif

typedef int v4i_vs __attribute__((vector_size(16)));
typedef __attribute__((address_space(1))) v4i_vs* gptr_b128;
typedef __attribute__((address_space(3))) v4i_vs* lptr_b128;

DEV void copy_b128_to_lds(const void* gsrc, void* ldst) {
#if HAVE_ASYNC_LDS
  __builtin_amdgcn_global_load_async_to_lds_b128(
      (gptr_b128)(uintptr_t)gsrc,
      (lptr_b128)(uint32_t)(uintptr_t)ldst,
      0, 0);
#else
  *(uint4*)ldst = *(const uint4*)gsrc;
#endif
}
DEV void async_copy_wait() {
#if HAVE_ASYNC_LDS
  __builtin_amdgcn_s_wait_asynccnt(0);
#endif
}

// ---- kernel 1a: weights fp32 -> bf16 ----------------------------------------
__global__ void convert_w_kernel(const float* __restrict__ Wz, const float* __restrict__ Wh,
                                 unsigned short* __restrict__ wzb, unsigned short* __restrict__ whb) {
  int i = blockIdx.x * blockDim.x + threadIdx.x;
  if (i < H_ * D_) {
    wzb[i] = f32_to_bf16(Wz[i]);
    whb[i] = f32_to_bf16(Wh[i]);
  }
}

// ---- kernel 1b: x fp32 -> bf16, 8 elems/thread, b128-coalesced --------------
__global__ void convert_x_kernel(const float* __restrict__ x, unsigned short* __restrict__ xb) {
  size_t gid = (size_t)blockIdx.x * blockDim.x + threadIdx.x;  // B*T*D/8 threads
  size_t i8 = gid * 8;
  float4 v0 = *(const float4*)(x + i8);
  float4 v1 = *(const float4*)(x + i8 + 4);
  uint4 o;
  o.x = pack_bf16x2(v0.x, v0.y);
  o.y = pack_bf16x2(v0.z, v0.w);
  o.z = pack_bf16x2(v1.x, v1.y);
  o.w = pack_bf16x2(v1.z, v1.w);
  *(uint4*)(xb + i8) = o;
}

// ---- kernel 2: dual bf16 WMMA GEMM + activations + fused chunk summaries ----
__launch_bounds__(256)
__global__ void gemm_act_kernel(const unsigned short* __restrict__ xb,
                                const unsigned short* __restrict__ wzb,
                                const unsigned short* __restrict__ whb,
                                const float* __restrict__ bz,
                                const float* __restrict__ bh,
                                float*  __restrict__ a_out,    // [B,T,H]  (ws)
                                float*  __restrict__ p_out,    // [B,T,H]  (d_out)
                                float2* __restrict__ seg_out)  // [B,NCHUNK,H]
{
  __shared__ unsigned short lA[2][TC * AStride];     // double-buffered A tile
  __shared__ unsigned short lB[2][2][NT * AStride];  // double-buffered {Wz,Wh}
  __shared__ float sA[8][2][NT];
  __shared__ float sP[8][2][NT];

  const int chunk = blockIdx.x;
  const int h0    = blockIdx.y * NT;
  const int bat   = blockIdx.z;
  const int t0    = chunk * TC;
  const int tid   = threadIdx.x;
  const int wave  = tid >> 5;
  const int lane  = tid & 31;
  const int lanec = lane & 15;
  const int half  = lane >> 4;
  const int mbase = wave * 16;

  const v8f vzero = {0.f,0.f,0.f,0.f,0.f,0.f,0.f,0.f};
  v8f acck[4], acch[4];
  for (int i = 0; i < 4; ++i) { acck[i] = vzero; acch[i] = vzero; }

  const size_t xrow = ((size_t)bat * T_ + t0) * (size_t)D_;

  // Stage one K-chunk (pure async b128 copies: A rows + both weight tiles).
  auto stage = [&](int kb, int buf) {
    // A: 128 rows x 32 bf16 = 4 x b128 per row -> 512 copies (2/thread)
    for (int i = tid; i < TC * 4; i += 256) {
      int row = i >> 2, q = i & 3;
      copy_b128_to_lds(xb + xrow + (size_t)row * D_ + kb + q * 8,
                       &lA[buf][row * AStride + q * 8]);
    }
    // B: 2 x 64 rows x 32 bf16 -> 512 copies (2/thread)
    for (int i = tid; i < 2 * NT * 4; i += 256) {
      int w = i >> 8, i2 = i & 255;
      int row = i2 >> 2, q = i2 & 3;
      const unsigned short* src = (w ? whb : wzb) + (size_t)(h0 + row) * D_ + kb + q * 8;
      copy_b128_to_lds(src, &lB[buf][w][row * AStride + q * 8]);
    }
  };

  stage(0, 0);

  const int mrow = mbase + lanec;
  const int kofA = half ? 16 : 0;
  const int kofB = half ? 16 : 0;

  for (int kc = 0; kc < D_ / KC; ++kc) {
    const int cur = kc & 1;
    async_copy_wait();      // drain this wave's copies for buffer `cur`
    __syncthreads();        // all waves staged `cur`; prev reads of `1-cur` done

    if (kc + 1 < D_ / KC) {
      stage((kc + 1) * KC, 1 - cur);  // overlaps with fragment loads + WMMAs
      if (kc + 2 < D_ / KC)           // speculative prefetch of k+2 A rows
        __builtin_prefetch(xb + xrow + (size_t)(tid >> 1) * D_ + (kc + 2) * KC, 0, 1);
    }

    // A fragment: lanes 0-15 K[0..15]; lanes 16-31 K[16..31]
    FragU af;
    af.q[0] = *(const uint4*)&lA[cur][mrow * AStride + kofA];
    af.q[1] = *(const uint4*)&lA[cur][mrow * AStride + kofA + 8];

    // Software-pipelined B fragments: issue loads for fragment i+1 before the
    // WMMA consuming fragment i, so only the older pair of ds_loads must drain
    // (partial s_wait_dscnt) and LDS latency overlaps WMMA issue.
    auto bptr = [&](int i) -> const unsigned short* {
      return &lB[cur][i & 1][(((i >> 1) * 16) + lanec) * AStride + kofB];
    };
    FragU fc, fn;
    fc.q[0] = *(const uint4*)bptr(0);
    fc.q[1] = *(const uint4*)(bptr(0) + 8);
#pragma unroll
    for (int i = 0; i < 8; ++i) {
      if (i < 7) {
        fn.q[0] = *(const uint4*)bptr(i + 1);
        fn.q[1] = *(const uint4*)(bptr(i + 1) + 8);
      }
      const int nt = i >> 1;
      if ((i & 1) == 0)
        acck[nt] = __builtin_amdgcn_wmma_f32_16x16x32_bf16(
            false, af.v, false, fc.v, (short)0, acck[nt], false, false);
      else
        acch[nt] = __builtin_amdgcn_wmma_f32_16x16x32_bf16(
            false, af.v, false, fc.v, (short)0, acch[nt], false, false);
      fc = fn;
    }
  }

  // Epilogue: bias + activations, write a/p, in-register chunk-scan partials.
  // C layout: VGPR r -> row M = r + 8*half, lane%16 -> col N.
  const int rowoff = half ? 8 : 0;
  for (int nt = 0; nt < 4; ++nt) {
    const int hcol = h0 + nt * 16 + lanec;
    const float bzv = bz[hcol];
    const float bhv = bh[hcol];
    float Arun = 1.f, Prun = 0.f;
    const size_t gbase = ((size_t)bat * T_ + (size_t)(t0 + mbase + rowoff)) * H_ + hcol;
    for (int r = 0; r < 8; ++r) {
      float kv = acck[nt][r] + bzv;
      float th = acch[nt][r] + bhv;
      float e  = __expf(-kv);
      float z  = 1.f / (1.f + e);   // sigmoid(k)
      float av = z * e;             // 1 - z == sigmoid(-k), stable
      float g;
      if (th >= 0.f) g = th + 0.5f;
      else { float eg = __expf(th); g = eg / (1.f + eg); }
      float pv = z * g;
      size_t gi = gbase + (size_t)r * H_;
      a_out[gi] = av;
      p_out[gi] = pv;
      Prun = av * Prun + pv;        // compose (Arun,Prun) with (av,pv)
      Arun = av * Arun;
    }
    sA[wave][half][nt * 16 + lanec] = Arun;
    sP[wave][half][nt * 16 + lanec] = Prun;
  }
  __syncthreads();

  // Reduce 16 partials (wave-major, half-minor == ascending t) per column.
  if (tid < NT) {
    float A = 1.f, P = 0.f;
    for (int w = 0; w < 8; ++w)
      for (int hf = 0; hf < 2; ++hf) {
        float Ai = sA[w][hf][tid], Pi = sP[w][hf][tid];
        P = Ai * P + Pi;
        A = Ai * A;
      }
    seg_out[((size_t)bat * NCHUNK + chunk) * H_ + (h0 + tid)] = make_float2(A, P);
  }
}

// ---- kernel 3: exclusive prefix over chunk summaries per (b,h) chain --------
__global__ void seg_prefix_kernel(const float2* __restrict__ seg,
                                  float* __restrict__ hstart) {
  int gid = blockIdx.x * blockDim.x + threadIdx.x;
  if (gid >= B_ * H_) return;
  int b = gid / H_, h = gid - b * H_;
  float hr = 0.f;  // h[-1] = 0
  for (int c = 0; c < NCHUNK; ++c) {
    size_t idx = ((size_t)b * NCHUNK + c) * H_ + h;
    hstart[idx] = hr;
    float2 s = seg[idx];
    hr = s.x * hr + s.y;
  }
}

// ---- kernel 4: apply recurrence within each chunk, overwrite p with h -------
__launch_bounds__(512)
__global__ void scan_apply_kernel(const float* __restrict__ a_in,
                                  const float* __restrict__ hstart,
                                  float* __restrict__ out) {
  const int chunk = blockIdx.x;
  const int bat   = blockIdx.y;
  const int h     = threadIdx.x;
  float hr = hstart[((size_t)bat * NCHUNK + chunk) * H_ + h];
  size_t idx = ((size_t)bat * T_ + (size_t)chunk * TC) * H_ + h;
  for (int t = 0; t < TC; ++t) {
    float av = a_in[idx];
    float pv = out[idx];   // p written by gemm_act this call
    hr = av * hr + pv;
    out[idx] = hr;
    idx += H_;
  }
}

extern "C" void kernel_launch(void* const* d_in, const int* in_sizes, int n_in,
                              void* d_out, int out_size, void* d_ws, size_t ws_size,
                              hipStream_t stream) {
  const float* x  = (const float*)d_in[0];
  const float* Wz = (const float*)d_in[1];
  const float* bz = (const float*)d_in[2];
  const float* Wh = (const float*)d_in[3];
  const float* bh = (const float*)d_in[4];
  float* out = (float*)d_out;

  char* ws = (char*)d_ws;
  size_t off = 0;
  unsigned short* wzb = (unsigned short*)(ws + off); off += (size_t)H_ * D_ * 2;  // 512 KB
  unsigned short* whb = (unsigned short*)(ws + off); off += (size_t)H_ * D_ * 2;  // 512 KB
  unsigned short* xb  = (unsigned short*)(ws + off); off += (size_t)B_ * T_ * D_ * 2;  // 67 MB
  float* a_ws = (float*)(ws + off);  off += (size_t)B_ * T_ * H_ * 4;             // 134 MB
  float2* seg = (float2*)(ws + off); off += (size_t)B_ * NCHUNK * H_ * 8;         // 2 MB
  float* hstart = (float*)(ws + off);                                             // 1 MB

  convert_w_kernel<<<(H_ * D_ + 255) / 256, 256, 0, stream>>>(Wz, Wh, wzb, whb);

  size_t n8 = (size_t)B_ * T_ * D_ / 8;
  convert_x_kernel<<<(unsigned)(n8 / 256), 256, 0, stream>>>(x, xb);

  dim3 g2(T_ / TC, H_ / NT, B_);  // 64 x 8 x 8 = 4096 blocks
  gemm_act_kernel<<<g2, 256, 0, stream>>>(xb, wzb, whb, bz, bh, a_ws, out, seg);

  seg_prefix_kernel<<<(B_ * H_ + 255) / 256, 256, 0, stream>>>(seg, hstart);

  dim3 g4(NCHUNK, B_);
  scan_apply_kernel<<<g4, H_, 0, stream>>>(a_ws, hstart, out);
}
